// BiLSTM_CRF_70033736728686
// MI455X (gfx1250) — matrix operand order repaired and checked by reading, not compiled, not run on hardware
//
#include <hip/hip_runtime.h>

// ---------------------------------------------------------------------------
// CRF NLL on gfx1250 (MI455X).
// Forward recurrence mapped onto v_wmma_f32_16x16x32_f16 via the rescaled
// linear-domain trick:  lin' = (E^T . lin) * exp(em_t),  E = exp(T - s).
// E^T is held as pre-formatted WMMA A-operands in registers for the whole scan.
// ---------------------------------------------------------------------------

#define B_SZ 512
#define L_SZ 2048
#define C_SZ 64
#define START_TAG 62
#define STOP_TAG 63

typedef __attribute__((ext_vector_type(16))) _Float16 v16h;
typedef __attribute__((ext_vector_type(8)))  _Float16 v8h;
typedef __attribute__((ext_vector_type(8)))  float    v8f;
typedef __attribute__((ext_vector_type(4)))  float    v4f;

// ---------------------------------------------------------------------------
// Kernel 1: precompute
//   s   = max(T)                      (global shift so E = exp(T-s) <= 1)
//   ET  = f16 E^T row-major [j][i]    (A-operand source: A[j,i] = exp(T[i,j]-s))
//   m2  = max_j T[j,STOP]
//   wstop[j] = exp(T[j,STOP]-m2)
// ---------------------------------------------------------------------------
__global__ void crf_prep(const float* __restrict__ T,
                         _Float16* __restrict__ ET,
                         float* __restrict__ wstop,
                         float* __restrict__ params) {
  __shared__ float red[256];
  const int tid = threadIdx.x;

  float mx = -3.4e38f;
  for (int k = tid; k < C_SZ * C_SZ; k += 256) mx = fmaxf(mx, T[k]);
  red[tid] = mx;
  __syncthreads();
  for (int w = 128; w > 0; w >>= 1) {
    if (tid < w) red[tid] = fmaxf(red[tid], red[tid + w]);
    __syncthreads();
  }
  const float s = red[0];
  __syncthreads();

  float m2p = -3.4e38f;
  if (tid < C_SZ) m2p = T[tid * C_SZ + STOP_TAG];
  red[tid] = m2p;
  __syncthreads();
  for (int w = 128; w > 0; w >>= 1) {
    if (tid < w) red[tid] = fmaxf(red[tid], red[tid + w]);
    __syncthreads();
  }
  const float m2 = red[0];
  __syncthreads();

  for (int k = tid; k < C_SZ * C_SZ; k += 256) {
    const int j = k >> 6, i = k & 63;           // ET[j][i] = exp(T[i][j]-s)
    ET[k] = (_Float16)__expf(T[i * C_SZ + j] - s);
  }
  if (tid < C_SZ) wstop[tid] = __expf(T[tid * C_SZ + STOP_TAG] - m2);
  if (tid == 0) { params[0] = s; params[1] = m2; }
}

// ---------------------------------------------------------------------------
// Kernel 2: gold path score, one wave per batch row, lanes stride over t.
// ---------------------------------------------------------------------------
__global__ void crf_score(const float* __restrict__ em,
                          const float* __restrict__ T,
                          const float* __restrict__ mask,
                          const int* __restrict__ tags,
                          float* __restrict__ scores) {
  const int lane = threadIdx.x & 31;
  const int b = blockIdx.x * (blockDim.x >> 5) + (threadIdx.x >> 5);
  if (b >= B_SZ) return;
  const float* emb = em   + (size_t)b * L_SZ * C_SZ;
  const int*   tgb = tags + (size_t)b * L_SZ;
  const float* mkb = mask + (size_t)b * L_SZ;

  float acc = 0.f, msum = 0.f;
  for (int t = lane; t < L_SZ; t += 32) {
    msum += mkb[t];
    if (t >= 1) {
      const int tg = tgb[t], tp = tgb[t - 1];
      acc += (emb[(size_t)t * C_SZ + tg] + T[tp * C_SZ + tg]) * mkb[t];
    }
  }
  for (int off = 16; off; off >>= 1) {
    acc  += __shfl_xor(acc,  off, 32);
    msum += __shfl_xor(msum, off, 32);
  }
  if (lane == 0) {
    const int t0 = tgb[0];
    float r = acc + emb[t0] + T[START_TAG * C_SZ + t0];
    const int last = (int)(msum + 0.5f) - 1;
    scores[b] = r + T[tgb[last] * C_SZ + STOP_TAG];
  }
}

// ---------------------------------------------------------------------------
// Kernel 3: forward scan. 1 wave per 16-batch column tile.
//   A tiles (constant)  : E^T, 4 M-tiles x 2 K-chunks, v16h each (64 VGPRs)
//   B operand (state)   : lin (f16), lane%16 = batch col, e = state half*16+e
//   D (f32)             : lane%16 = batch col, VGPR r = state half*8+r
// Per step: 8 WMMA -> *exp(em) -> column max (shfl_xor 16) -> rescale ->
//           butterfly D->B layout swap -> f16.
// ---------------------------------------------------------------------------
__global__ __launch_bounds__(32) void crf_forward(
    const float* __restrict__ em, const float* __restrict__ mask,
    const float* __restrict__ Tr, const _Float16* __restrict__ ET,
    const float* __restrict__ wstop, const float* __restrict__ params,
    float* __restrict__ partition) {
  const int lane = threadIdx.x;
  const int lm   = lane & 15;
  const int half = lane >> 4;
  const int b0   = blockIdx.x * 16;
  const float s_shift = params[0];

  // ---- constant A operands: two contiguous 16B loads per v16h ----
  v16h A[4][2];
#pragma unroll
  for (int jT = 0; jT < 4; ++jT)
#pragma unroll
    for (int c = 0; c < 2; ++c) {
      const _Float16* p = ET + (jT * 16 + lm) * C_SZ + c * 32 + half * 16;
      v8h lo = *(const v8h*)(p);
      v8h hi = *(const v8h*)(p + 8);
#pragma unroll
      for (int e = 0; e < 8; ++e) { A[jT][c][e] = lo[e]; A[jT][c][8 + e] = hi[e]; }
    }

  const float* emB = em + (size_t)(b0 + lm) * L_SZ * C_SZ;  // this lane's batch column
  const float* mkB = mask + (size_t)(b0 + lm) * L_SZ;

  // ---- init: alpha0 = em[:,0,:] + T[START,:]; to rescaled linear f16 ----
  float ls = 0.f;
  v16h Bv[2];
  {
    float x[2][16];
    float m = -3.4e38f;
#pragma unroll
    for (int c = 0; c < 2; ++c) {
      const int jb = c * 32 + half * 16;
#pragma unroll
      for (int e = 0; e < 16; e += 4) {
        v4f ev = *(const v4f*)(emB + jb + e);
        v4f tv = *(const v4f*)(Tr + START_TAG * C_SZ + jb + e);
#pragma unroll
        for (int q = 0; q < 4; ++q) {
          x[c][e + q] = ev[q] + tv[q];
          m = fmaxf(m, x[c][e + q]);
        }
      }
    }
    m = fmaxf(m, __shfl_xor(m, 16, 32));
    ls = m;
#pragma unroll
    for (int c = 0; c < 2; ++c)
#pragma unroll
      for (int e = 0; e < 16; ++e) Bv[c][e] = (_Float16)__expf(x[c][e] - m);
  }

  // ---- sequential scan over t ----
  for (int t = 1; t < L_SZ; ++t) {
    // emissions for this step, D-layout: state j = jT*16 + half*8 + r
    v4f emv[4][2];
    const float* pe = emB + (size_t)t * C_SZ + half * 8;
#pragma unroll
    for (int jT = 0; jT < 4; ++jT) {
      emv[jT][0] = *(const v4f*)(pe + jT * 16);
      emv[jT][1] = *(const v4f*)(pe + jT * 16 + 4);
    }
    const float mk = mkB[t];

    // 8 chained WMMAs: D[jT] = A[jT][0]*B[0] + A[jT][1]*B[1]
    v8f acc[4];
#pragma unroll
    for (int jT = 0; jT < 4; ++jT) {
      v8f z = {};
      z = __builtin_amdgcn_wmma_f32_16x16x32_f16(false, A[jT][0], false, Bv[0],
                                                 (short)0, z, false, false);
      acc[jT] = __builtin_amdgcn_wmma_f32_16x16x32_f16(false, A[jT][1], false, Bv[1],
                                                       (short)0, z, false, false);
    }

    // * exp(em) and per-batch-column max
    float x[4][8];
    float m = -3.4e38f;
#pragma unroll
    for (int jT = 0; jT < 4; ++jT)
#pragma unroll
      for (int r = 0; r < 8; ++r) {
        const float e = (r < 4) ? emv[jT][0][r] : emv[jT][1][r - 4];
        const float v = acc[jT][r] * __expf(e);
        x[jT][r] = v;
        m = fmaxf(m, v);
      }
    m = fmaxf(m, __shfl_xor(m, 16, 32));
    const float inv = 1.0f / m;

    // D -> next B layout: butterfly swap across lane halves (N = lane%16 fixed)
    v16h Bn[2];
#pragma unroll
    for (int c = 0; c < 2; ++c) {
#pragma unroll
      for (int r = 0; r < 8; ++r) {
        const float own = half ? x[c * 2 + 1][r] : x[c * 2][r];
        const float oth = half ? x[c * 2][r]     : x[c * 2 + 1][r];
        const float par = __shfl_xor(oth, 16, 32);
        const _Float16 ho = (_Float16)(own * inv);
        const _Float16 hp = (_Float16)(par * inv);
        Bn[c][r]     = half ? hp : ho;
        Bn[c][8 + r] = half ? ho : hp;
      }
    }
    const float lsn = ls + __logf(m) + s_shift;
    const bool upd = mk > 0.f;     // mask: keep old alpha where mask==0
#pragma unroll
    for (int c = 0; c < 2; ++c)
#pragma unroll
      for (int e = 0; e < 16; ++e) Bv[c][e] = upd ? Bn[c][e] : Bv[c][e];
    ls = upd ? lsn : ls;
  }

  // ---- partition[b] = log(sum_j lin_j * wstop_j) + m2 + ls ----
  const float m2 = params[1];
  float tot = 0.f;
#pragma unroll
  for (int c = 0; c < 2; ++c) {
    const int jb = c * 32 + half * 16;
#pragma unroll
    for (int e = 0; e < 16; e += 4) {
      v4f wv = *(const v4f*)(wstop + jb + e);
#pragma unroll
      for (int q = 0; q < 4; ++q) tot += (float)Bv[c][e + q] * wv[q];
    }
  }
  tot += __shfl_xor(tot, 16, 32);
  if (half == 0) partition[b0 + lm] = __logf(tot) + m2 + ls;
}

// ---------------------------------------------------------------------------
// Kernel 4: mean(partition - scores) -> scalar
// ---------------------------------------------------------------------------
__global__ void crf_finish(const float* __restrict__ partition,
                           const float* __restrict__ scores,
                           float* __restrict__ out) {
  const int lane = threadIdx.x;
  float acc = 0.f;
  for (int b = lane; b < B_SZ; b += 32) acc += partition[b] - scores[b];
  for (int off = 16; off; off >>= 1) acc += __shfl_xor(acc, off, 32);
  if (lane == 0) out[0] = acc / (float)B_SZ;
}

// ---------------------------------------------------------------------------
extern "C" void kernel_launch(void* const* d_in, const int* in_sizes, int n_in,
                              void* d_out, int out_size, void* d_ws, size_t ws_size,
                              hipStream_t stream) {
  const float* em   = (const float*)d_in[0];   // [512,2048,64] f32
  const float* T    = (const float*)d_in[1];   // [64,64] f32
  const float* mask = (const float*)d_in[2];   // [512,2048] f32
  const int*   tags = (const int*)d_in[3];     // [512,2048] i32
  float* out = (float*)d_out;

  char* ws = (char*)d_ws;
  _Float16* ET     = (_Float16*)(ws + 0);      // 8192 B
  float*    wstop  = (float*)(ws + 8192);      //  256 B
  float*    params = (float*)(ws + 8448);      //   64 B
  float*    scores = (float*)(ws + 8512);      // 2048 B
  float*    part   = (float*)(ws + 10560);     // 2048 B

  crf_prep<<<1, 256, 0, stream>>>(T, ET, wstop, params);
  crf_score<<<B_SZ / 8, 256, 0, stream>>>(em, T, mask, tags, scores);
  crf_forward<<<B_SZ / 16, 32, 0, stream>>>(em, mask, T, ET, wstop, params, part);
  crf_finish<<<1, 32, 0, stream>>>(part, scores, out);
}
